// SelfAttention2D_diffusion_31095563223730
// MI455X (gfx1250) — matrix operand order repaired and checked by reading, not compiled, not run on hardware
//
#include <hip/hip_runtime.h>
#include <hip/hip_bf16.h>

typedef __attribute__((ext_vector_type(16))) _Float16 v16h;
typedef __attribute__((ext_vector_type(8)))  _Float16 v8h;
typedef __attribute__((ext_vector_type(4)))  _Float16 v4h;
typedef __attribute__((ext_vector_type(2)))  _Float16 v2h;
typedef __attribute__((ext_vector_type(2)))  __fp16   v2fp16;   // cvt_pkrtz native type
typedef __attribute__((ext_vector_type(8)))  float    v8f;
typedef __attribute__((ext_vector_type(4)))  float    v4f;

#define C_DIM   256
#define NHEAD   8
#define DHEAD   32
#define NTOK    4096
#define BATCH   2
#define QKV_N   (3 * C_DIM)
#define NROWS   (BATCH * NTOK)         // 8192
#define LN_EPS  1e-5f
// d^-0.5 * log2(e): scores land in log2 domain -> softmax uses raw v_exp_f32
#define QSCALE_L2E (0.17677669529663687f * 1.4426950408889634f)
#define KTILE   64                     // keys per flash iteration
#define VT_LD   72                     // padded vt row stride (halves); 144B rows, 16B-aligned chunks

// ---------------------------------------------------------------------------
// helpers
// ---------------------------------------------------------------------------
__device__ __forceinline__ v16h load_frag(const _Float16* p0, const _Float16* p1) {
    v16h r;
    *(v8h*)&r       = *(const v8h*)p0;
    *((v8h*)&r + 1) = *(const v8h*)p1;
    return r;
}

__device__ __forceinline__ v8f wmma_f16(v16h a, v16h b, v8f c) {
    return __builtin_amdgcn_wmma_f32_16x16x32_f16(false, a, false, b, (short)0, c,
                                                  false, false);
}

__device__ __forceinline__ v2h cvt_pk_f16(float a, float b) {
    v2fp16 r = __builtin_amdgcn_cvt_pkrtz(a, b);
    return *(v2h*)&r;   // bit-identical reinterpret
}

// xor-MASK butterfly via immediate-operand LDS swizzle (no address VGPR)
template <int MASK>
__device__ __forceinline__ float swz_xor(float x) {
    return __int_as_float(
        __builtin_amdgcn_ds_swizzle(__float_as_int(x), 0x1f | (MASK << 10)));
}

// online-softmax update over one 16-query x 64-key S^T tile held per-lane.
// s[] is consumed; pk[] receives P (f16, packed pairwise) in A-frag chunk layout.
__device__ __forceinline__ float softmax_update(v8f s[4], float& m, float& l,
                                                v8h pk[4]) {
    float mx = fmaxf(s[0][0], s[0][1]);
#pragma unroll
    for (int e = 2; e < 8; ++e) mx = fmaxf(mx, s[0][e]);
#pragma unroll
    for (int sub = 1; sub < 4; ++sub)
#pragma unroll
        for (int e = 0; e < 8; ++e) mx = fmaxf(mx, s[sub][e]);
    mx = fmaxf(mx, swz_xor<16>(mx));

    float mn  = fmaxf(m, mx);
    float esc = __builtin_amdgcn_exp2f(m - mn);
    float rs  = 0.0f;
#pragma unroll
    for (int sub = 0; sub < 4; ++sub) {
        v2h* pp = (v2h*)&pk[sub];
#pragma unroll
        for (int i = 0; i < 4; ++i) {
            float p0 = __builtin_amdgcn_exp2f(s[sub][2 * i]     - mn);
            float p1 = __builtin_amdgcn_exp2f(s[sub][2 * i + 1] - mn);
            rs += p0 + p1;
            pp[i] = cvt_pk_f16(p0, p1);
        }
    }
    rs += swz_xor<16>(rs);
    l = l * esc + rs;
    m = mn;
    return esc;
}

// ---------------------------------------------------------------------------
// 1) AdaLN conditioning: ss[b, 0:512] = silu(emb_table[t[b]]) @ ada_w + ada_b
// ---------------------------------------------------------------------------
__global__ __launch_bounds__(256) void ada_kernel(const float* __restrict__ emb_table,
                                                  const int* __restrict__ t,
                                                  const float* __restrict__ ada_w,
                                                  const float* __restrict__ ada_b,
                                                  float* __restrict__ ss) {
    int b   = blockIdx.x;
    int col = blockIdx.y * 256 + threadIdx.x;   // 0..511
    const float* e = emb_table + t[b] * C_DIM;
    float acc = ada_b[col];
#pragma unroll 4
    for (int c = 0; c < C_DIM; ++c) {
        float v = e[c];
        float s = v / (1.0f + __expf(-v));
        acc += s * ada_w[c * (2 * C_DIM) + col];
    }
    ss[b * (2 * C_DIM) + col] = acc;
}

// ---------------------------------------------------------------------------
// 2) fp32 -> fp16 weight conversion
// ---------------------------------------------------------------------------
__global__ __launch_bounds__(256) void cvt_kernel(const float* __restrict__ src,
                                                  _Float16* __restrict__ dst, int n) {
    int i = blockIdx.x * 256 + threadIdx.x;
    if (i < n) dst[i] = (_Float16)src[i];
}

// ---------------------------------------------------------------------------
// 3) LayerNorm + AdaLN modulation, fp16 output. One WAVE per token: 8 floats
//    per lane, swizzle-butterfly reduction, no LDS, no barriers.
// ---------------------------------------------------------------------------
__global__ __launch_bounds__(256) void lnmod_kernel(const float* __restrict__ x,
                                                    const float* __restrict__ ss,
                                                    _Float16* __restrict__ xmod) {
    int wave = threadIdx.x >> 5;
    int lane = threadIdx.x & 31;
    int tok  = blockIdx.x * 8 + wave;       // 0..8191
    int b    = tok >> 12;
    int c0   = lane * 8;

    const float* xp = x + (size_t)tok * C_DIM + c0;
    v4f x0 = *(const v4f*)xp;
    v4f x1 = *(const v4f*)(xp + 4);

    float sum = ((x0[0] + x0[1]) + (x0[2] + x0[3])) +
                ((x1[0] + x1[1]) + (x1[2] + x1[3]));
    float sq  = x0[0]*x0[0] + x0[1]*x0[1] + x0[2]*x0[2] + x0[3]*x0[3] +
                x1[0]*x1[0] + x1[1]*x1[1] + x1[2]*x1[2] + x1[3]*x1[3];

    sum += swz_xor<1>(sum);   sq += swz_xor<1>(sq);
    sum += swz_xor<2>(sum);   sq += swz_xor<2>(sq);
    sum += swz_xor<4>(sum);   sq += swz_xor<4>(sq);
    sum += swz_xor<8>(sum);   sq += swz_xor<8>(sq);
    sum += swz_xor<16>(sum);  sq += swz_xor<16>(sq);

    float mu  = sum * (1.0f / C_DIM);
    float var = sq * (1.0f / C_DIM) - mu * mu;
    float rstd = rsqrtf(var + LN_EPS);

    const float* scp = ss + b * (2 * C_DIM) + c0;
    const float* shp = scp + C_DIM;
    v4f sc0 = *(const v4f*)scp,       sc1 = *(const v4f*)(scp + 4);
    v4f sh0 = *(const v4f*)shp,       sh1 = *(const v4f*)(shp + 4);

    v8h o;
#pragma unroll
    for (int i = 0; i < 4; ++i) {
        o[i]     = (_Float16)(((x0[i] - mu) * rstd) * (1.0f + sc0[i]) + sh0[i]);
        o[4 + i] = (_Float16)(((x1[i] - mu) * rstd) * (1.0f + sc1[i]) + sh1[i]);
    }
    *(v8h*)(xmod + (size_t)tok * C_DIM + c0) = o;
}

// ---------------------------------------------------------------------------
// 4/6) WMMA GEMM: out[M=8192, Ncols] = A[8192, 256] @ W[256, Ncols] + bias
//   MODE 0: Ncols=768, scatter into head-major q/k/v (q pre-scaled, log2e folded)
//   MODE 1: Ncols=256, plain fp32 store
// Block = 256 threads (8 waves). Tile: M=128 (16 rows/wave), N=64, K step 32.
// ---------------------------------------------------------------------------
template <int MODE>
__global__ __launch_bounds__(256) void gemm_kernel(const _Float16* __restrict__ A,
                                                   const _Float16* __restrict__ W,
                                                   const float* __restrict__ bias,
                                                   _Float16* __restrict__ qs,
                                                   _Float16* __restrict__ ks,
                                                   _Float16* __restrict__ vs,
                                                   float* __restrict__ fout,
                                                   int Ncols) {
    __shared__ _Float16 bt[64 * 32];   // B tile, transposed: [col][k]

    int tid  = threadIdx.x;
    int wave = tid >> 5;
    int lane = tid & 31;
    int m0   = blockIdx.x * 128 + wave * 16;
    int n0   = blockIdx.y * 64;
    int hsel = (lane < 16) ? 0 : 8;    // C-frag row offset (epilogue)
    int koff = (lane < 16) ? 0 : 16;   // B-frag K chunk select
    int row  = m0 + (lane & 15);

    // staging indices: thread covers 4 cols x 2 k-rows, packed b32 LDS stores
    int cc0 = (tid & 15) * 4;          // 0..60
    int kk2 = (tid >> 4) * 2;          // 0..30

    v8f acc[4] = {};

    for (int k0 = 0; k0 < C_DIM; k0 += 32) {
        __syncthreads();   // protect previous iteration's bt reads
        {
            const _Float16* wp = W + (size_t)(k0 + kk2) * Ncols + n0 + cc0;
            v4h wa = *(const v4h*)wp;
            v4h wb = *(const v4h*)(wp + Ncols);
#pragma unroll
            for (int i = 0; i < 4; ++i) {
                v2h pr; pr[0] = wa[i]; pr[1] = wb[i];
                *(v2h*)&bt[(cc0 + i) * 32 + kk2] = pr;
            }
        }
        __syncthreads();

        const _Float16* ap = A + row * C_DIM + k0;
        __builtin_prefetch(ap + 32, 0, 0);
        v16h af = load_frag(ap + koff, ap + koff + 8);

#pragma unroll
        for (int n = 0; n < 4; ++n) {
            const _Float16* bp = &bt[(n * 16 + (lane & 15)) * 32 + koff];
            v16h bf = load_frag(bp, bp + 8);
            acc[n] = wmma_f16(af, bf, acc[n]);
        }
    }

    // epilogue: C layout -> row = m0 + e + hsel, col = n0 + n*16 + (lane&15)
#pragma unroll
    for (int n = 0; n < 4; ++n) {
        int col = n0 + n * 16 + (lane & 15);
        float bv = bias[col];
#pragma unroll
        for (int e = 0; e < 8; ++e) {
            int r = m0 + e + hsel;
            float val = acc[n][e] + bv;
            if (MODE == 0) {
                int s  = col >> 8;            // 0=q 1=k 2=v
                int hh = (col >> 5) & 7;
                int dd = col & 31;
                int b  = r >> 12;
                int nr = r & (NTOK - 1);
                if (s == 0) val *= QSCALE_L2E;
                _Float16* dst = (s == 0) ? qs : ((s == 1) ? ks : vs);
                dst[((b * NHEAD + hh) * NTOK + nr) * DHEAD + dd] = (_Float16)val;
            } else {
                fout[r * Ncols + col] = val;
            }
        }
    }
}

// ---------------------------------------------------------------------------
// 5) Flash attention, S-transposed, 2 query tiles per wave.
//    Grid: (B*h, N/128). Block = 128 threads (4 waves), 32 queries per wave.
//    Per 64-key iteration and wave: 8 S WMMAs + 8 O WMMAs; K fragments and V
//    fragments each reused across both query tiles. O-rescale skipped unless a
//    new running max appeared (wave-coherent branch).
// ---------------------------------------------------------------------------
__global__ __launch_bounds__(128) void attn_kernel(const _Float16* __restrict__ qs,
                                                   const _Float16* __restrict__ ks,
                                                   const _Float16* __restrict__ vs,
                                                   _Float16* __restrict__ ao) {
    __shared__ _Float16 vt[DHEAD * VT_LD];  // V tile transposed: [d][key], padded

    int tid  = threadIdx.x;
    int wave = tid >> 5;
    int lane = tid & 31;
    int l15  = lane & 15;
    int bh   = blockIdx.x;                  // 0..15
    int b    = bh >> 3;
    int hh   = bh & 7;
    int qa   = blockIdx.y * 128 + wave * 32;   // query tile a
    int qb   = qa + 16;                        // query tile b
    int hsel = (lane < 16) ? 0 : 8;
    int koff = (lane < 16) ? 0 : 16;

    // Q^T B-fragments: lane = query column, 16 contiguous d per lane chunk
    const _Float16* qra = qs + ((size_t)bh * NTOK + qa + l15) * DHEAD;
    v16h qf0 = load_frag(qra + koff, qra + koff + 8);
    const _Float16* qrb = qra + 16 * DHEAD;
    v16h qf1 = load_frag(qrb + koff, qrb + koff + 8);

    v8f oa0 = {}, oa1 = {}, ob0 = {}, ob1 = {};
    float ma = -3.0e38f, la = 0.0f;
    float mb = -3.0e38f, lb = 0.0f;

    const _Float16* kbase = ks + (size_t)bh * NTOK * DHEAD;
    const _Float16* vbase = vs + (size_t)bh * NTOK * DHEAD;

    // V staging: thread covers 2 key rows x 8 d -> packed b32 LDS stores
    int vr2 = (tid & 31) * 2;          // key row pair 0..62
    int vd0 = (tid >> 5) * 8;          // d chunk 0,8,16,24

    for (int j0 = 0; j0 < NTOK; j0 += KTILE) {
        // ---- stage V^T into LDS (issue early; overlaps S WMMA + softmax)
        __syncthreads();   // previous iteration's vt consumers done
        {
            const _Float16* vp = vbase + (size_t)(j0 + vr2) * DHEAD + vd0;
            __builtin_prefetch(vp + KTILE * DHEAD, 0, 0);
            v8h va = *(const v8h*)vp;
            v8h vb = *(const v8h*)(vp + DHEAD);
#pragma unroll
            for (int i = 0; i < 8; ++i) {
                v2h pr; pr[0] = va[i]; pr[1] = vb[i];
                *(v2h*)&vt[(vd0 + i) * VT_LD + vr2] = pr;
            }
        }

        // ---- S^T tiles for both query tiles; K fragment loaded once, used twice
        v8f sa[4], sb[4];
#pragma unroll
        for (int sub = 0; sub < 4; ++sub) {
            const _Float16* krow = kbase + (size_t)(j0 + sub * 16 + l15) * DHEAD;
            v16h kf = load_frag(krow + koff, krow + koff + 8);
            v8f z = {};
            sa[sub] = wmma_f16(kf, qf0, z);
            sb[sub] = wmma_f16(kf, qf1, z);
        }

        // ---- online softmax per query tile (lane == query column)
        v8h pka[4], pkb[4];
        float esca = softmax_update(sa, ma, la, pka);
        float escb = softmax_update(sb, mb, lb, pkb);

        // rescale O accumulators only when a new running max appeared
        if (__any((esca < 1.0f) | (escb < 1.0f))) {
#pragma unroll
            for (int e = 0; e < 8; ++e) {
                float fa = __shfl(esca, e + hsel, 32);
                float fb = __shfl(escb, e + hsel, 32);
                oa0[e] *= fa; oa1[e] *= fa;
                ob0[e] *= fb; ob1[e] *= fb;
            }
        }

        __syncthreads();   // vt writes visible

        // ---- O += P x V : A = P (registers), B = V^T chunks from LDS (reused)
#pragma unroll
        for (int c = 0; c < 2; ++c) {
            const _Float16* vp0 = &vt[l15 * VT_LD + c * 32 + hsel];
            const _Float16* vp1 = &vt[(16 + l15) * VT_LD + c * 32 + hsel];
            v16h vf0 = load_frag(vp0, vp0 + 16);
            v16h vf1 = load_frag(vp1, vp1 + 16);

            v16h pfa;
            *(v8h*)&pfa       = pka[2 * c];
            *((v8h*)&pfa + 1) = pka[2 * c + 1];
            oa0 = wmma_f16(pfa, vf0, oa0);
            oa1 = wmma_f16(pfa, vf1, oa1);

            v16h pfb;
            *(v8h*)&pfb       = pkb[2 * c];
            *((v8h*)&pfb + 1) = pkb[2 * c + 1];
            ob0 = wmma_f16(pfb, vf0, ob0);
            ob1 = wmma_f16(pfb, vf1, ob1);
        }
    }

    // ---- finalize: divide rows by l, scatter to [B, N, C] fp16
    float inva = __builtin_amdgcn_rcpf(la);
    float invb = __builtin_amdgcn_rcpf(lb);
#pragma unroll
    for (int e = 0; e < 8; ++e) {
        float fa = __shfl(inva, e + hsel, 32);
        float fb = __shfl(invb, e + hsel, 32);
        int ra = qa + e + hsel;
        int rb = qb + e + hsel;
        _Float16* da = ao + ((size_t)(b * NTOK + ra)) * C_DIM + hh * DHEAD;
        _Float16* db = ao + ((size_t)(b * NTOK + rb)) * C_DIM + hh * DHEAD;
        da[l15]      = (_Float16)(oa0[e] * fa);
        da[16 + l15] = (_Float16)(oa1[e] * fa);
        db[l15]      = (_Float16)(ob0[e] * fb);
        db[16 + l15] = (_Float16)(ob1[e] * fb);
    }
}

// ---------------------------------------------------------------------------
// launch
// ---------------------------------------------------------------------------
static inline size_t align_up(size_t v, size_t a) { return (v + a - 1) & ~(a - 1); }

extern "C" void kernel_launch(void* const* d_in, const int* in_sizes, int n_in,
                              void* d_out, int out_size, void* d_ws, size_t ws_size,
                              hipStream_t stream) {
    const float* x         = (const float*)d_in[0];
    // d_in[1] = cond (unused by reference)
    const int*   t         = (const int*)d_in[2];
    const float* emb_table = (const float*)d_in[3];
    const float* ada_w     = (const float*)d_in[4];
    const float* ada_b     = (const float*)d_in[5];
    const float* qkv_w     = (const float*)d_in[6];
    const float* qkv_b     = (const float*)d_in[7];
    const float* proj_w    = (const float*)d_in[8];
    const float* proj_b    = (const float*)d_in[9];
    (void)in_sizes; (void)n_in; (void)out_size; (void)ws_size;

    char* w = (char*)d_ws;
    size_t off = 0;
    float* ss = (float*)(w + off);          off = align_up(off + BATCH * 2 * C_DIM * 4, 256);
    _Float16* wqkv  = (_Float16*)(w + off); off = align_up(off + C_DIM * QKV_N * 2, 256);
    _Float16* wproj = (_Float16*)(w + off); off = align_up(off + C_DIM * C_DIM * 2, 256);
    _Float16* xmod  = (_Float16*)(w + off); off = align_up(off + (size_t)NROWS * C_DIM * 2, 256);
    _Float16* qsb   = (_Float16*)(w + off); off = align_up(off + (size_t)NROWS * C_DIM * 2, 256);
    _Float16* ksb   = (_Float16*)(w + off); off = align_up(off + (size_t)NROWS * C_DIM * 2, 256);
    _Float16* vsb   = (_Float16*)(w + off); off = align_up(off + (size_t)NROWS * C_DIM * 2, 256);
    _Float16* ao    = (_Float16*)(w + off); off = align_up(off + (size_t)NROWS * C_DIM * 2, 256);

    // 1) conditioning
    ada_kernel<<<dim3(BATCH, 2), 256, 0, stream>>>(emb_table, t, ada_w, ada_b, ss);

    // 2) weight conversion
    cvt_kernel<<<(C_DIM * QKV_N + 255) / 256, 256, 0, stream>>>(qkv_w, wqkv, C_DIM * QKV_N);
    cvt_kernel<<<(C_DIM * C_DIM + 255) / 256, 256, 0, stream>>>(proj_w, wproj, C_DIM * C_DIM);

    // 3) LN + modulation (wave-per-token, barrier-free)
    lnmod_kernel<<<NROWS / 8, 256, 0, stream>>>(x, ss, xmod);

    // 4) QKV projection (WMMA), scatter head-major
    gemm_kernel<0><<<dim3(NROWS / 128, QKV_N / 64), 256, 0, stream>>>(
        xmod, wqkv, qkv_b, qsb, ksb, vsb, nullptr, QKV_N);

    // 5) flash attention (WMMA, S-transposed, dual query tiles)
    attn_kernel<<<dim3(BATCH * NHEAD, NTOK / 128), 128, 0, stream>>>(qsb, ksb, vsb, ao);

    // 6) output projection (WMMA) -> fp32 d_out
    gemm_kernel<1><<<dim3(NROWS / 128, C_DIM / 64), 256, 0, stream>>>(
        ao, wproj, proj_b, nullptr, nullptr, nullptr, (float*)d_out, C_DIM);
}
